// CrossAttention_31026843746450
// MI455X (gfx1250) — matrix operand order repaired and checked by reading, not compiled, not run on hardware
//
#include <hip/hip_runtime.h>
#include <hip/hip_bf16.h>
#include <stdint.h>

// ---------------------------------------------------------------------------
// CrossAttention (B=2, C=64, N=4096, Cinner=8) — fused flash-attention on
// CDNA5 WMMA (v_wmma_f32_16x16x32_f16), wave32.
//
// Round-1 change: m-rows of the energy tile are permuted at load time so the
// C-layout E tile lands exactly in B-matrix K order => the Pt operand for the
// PV WMMA is built with zero cross-lane data movement. The two remaining
// softmax half-combines use v_permlanex16_b32 (VALU) instead of ds_bpermute.
// ---------------------------------------------------------------------------

#define BATCH 2
#define CH    64
#define C8    8
#define NTOK  4096   // 16*16*16

typedef __attribute__((ext_vector_type(16))) _Float16 v16h;
typedef __attribute__((ext_vector_type(8)))  float    v8f;

union V16 { v16h v; uint32_t u[8]; };

static __device__ __forceinline__ uint16_t f2h(float f) {
    _Float16 h = (_Float16)f;
    return __builtin_bit_cast(uint16_t, h);
}
static __device__ __forceinline__ uint32_t pkh(float lo, float hi) {
    return (uint32_t)f2h(lo) | ((uint32_t)f2h(hi) << 16);
}
// Swap the two 16-lane halves of a wave32 (lane i <-> lane i^16) in VALU:
// v_permlanex16_b32 with identity nibble selects.
static __device__ __forceinline__ float swap16f(float v) {
    const int s = __builtin_bit_cast(int, v);
    const int r = __builtin_amdgcn_permlanex16(
        s, s, 0x76543210, (int)0xfedcba98, false, false);
    return __builtin_bit_cast(float, r);
}

// ---------------------------------------------------------------------------
// Kernel 1: q/k/v projections.
//   qpk, kpk : uint32 [B][4][N], word j at col n = pack_f16(q[2j][n], q[2j+1][n])
//   vh       : f16    [B][64][N]
// ---------------------------------------------------------------------------
__global__ __launch_bounds__(256)
void ca_proj_kernel(const float* __restrict__ x, const float* __restrict__ ctx,
                    const float* __restrict__ Wq, const float* __restrict__ bq,
                    const float* __restrict__ Wk, const float* __restrict__ bk,
                    const float* __restrict__ Wv, const float* __restrict__ bv,
                    uint32_t* __restrict__ qpk, uint32_t* __restrict__ kpk,
                    uint16_t* __restrict__ vh)
{
    __shared__ float sWq[C8 * CH];
    __shared__ float sWk[C8 * CH];
    __shared__ float sWv[CH * CH];
    __shared__ float sb[2 * C8 + CH];   // [0..7]=bq [8..15]=bk [16..79]=bv

    const int tid = threadIdx.x;
    for (int i = tid; i < C8 * CH; i += 256) { sWq[i] = Wq[i]; sWk[i] = Wk[i]; }
    for (int i = tid; i < CH * CH; i += 256) sWv[i] = Wv[i];
    if (tid < C8)               sb[tid] = bq[tid];
    else if (tid < 2 * C8)      sb[tid] = bk[tid - C8];
    else if (tid < 2 * C8 + CH) sb[tid] = bv[tid - 2 * C8];
    __syncthreads();

    const int idx = blockIdx.x * 256 + tid;
    const int b   = idx >> 12;          // / NTOK
    const int n   = idx & (NTOK - 1);

    float col[CH];

    // ---- context column -> k, v ----
#pragma unroll
    for (int c = 0; c < CH; ++c)
        col[c] = ctx[(size_t)(b * CH + c) * NTOK + n];

    float kk[C8];
#pragma unroll
    for (int j = 0; j < C8; ++j) {
        float a = sb[C8 + j];
#pragma unroll
        for (int c = 0; c < CH; ++c) a += sWk[j * CH + c] * col[c];
        kk[j] = a;
    }
#pragma unroll
    for (int j = 0; j < 4; ++j)
        kpk[(size_t)(b * 4 + j) * NTOK + n] = pkh(kk[2 * j], kk[2 * j + 1]);

    for (int co = 0; co < CH; ++co) {
        float a = sb[2 * C8 + co];
#pragma unroll
        for (int c = 0; c < CH; ++c) a += sWv[co * CH + c] * col[c];
        vh[(size_t)(b * CH + co) * NTOK + n] = f2h(a);
    }

    // ---- x column -> q ----
#pragma unroll
    for (int c = 0; c < CH; ++c)
        col[c] = x[(size_t)(b * CH + c) * NTOK + n];

    float qq[C8];
#pragma unroll
    for (int j = 0; j < C8; ++j) {
        float a = sb[j];
#pragma unroll
        for (int c = 0; c < CH; ++c) a += sWq[j * CH + c] * col[c];
        qq[j] = a;
    }
#pragma unroll
    for (int j = 0; j < 4; ++j)
        qpk[(size_t)(b * 4 + j) * NTOK + n] = pkh(qq[2 * j], qq[2 * j + 1]);
}

// ---------------------------------------------------------------------------
// Kernel 2: fused flash attention + gamma residual.
// grid = B * N/16 blocks, 128 threads (4 waves). Each wave owns 16 channels
// of v/out; energy tile + softmax stats are computed redundantly per wave
// (inner dim is only 8) so waves never communicate.
//
// m-row permutation: tile0 rows {0..7, 8..15} -> m0 + {0..7, 16..23},
//                    tile1 rows {0..7, 8..15} -> m0 + {8..15, 24..31}.
// => E C-layout half h holds m0+16h+{0..15} per lane, exactly the K range
//    the B-matrix half needs for the PV WMMA: Pt is packed locally.
// ---------------------------------------------------------------------------
__global__ __launch_bounds__(128)
void ca_attn_kernel(const uint32_t* __restrict__ qpk, const uint32_t* __restrict__ kpk,
                    const uint16_t* __restrict__ vh, const float* __restrict__ x,
                    const float* __restrict__ gamma_p, float* __restrict__ out)
{
    const int lane  = threadIdx.x & 31;
    const int wave  = threadIdx.x >> 5;
    const int ln    = lane & 15;     // column / row-in-tile index
    const int hh    = lane >> 4;     // lane half
    const int b     = blockIdx.x >> 8;              // / (NTOK/16)
    const int n0    = (blockIdx.x & 255) << 4;
    const int n     = n0 + ln;
    const int cbase = wave << 4;

    const float gamma = gamma_p[0];

    // Q as B-matrix (K = channel, only 0..7 valid; lanes>=16 carry K=16..31 -> 0)
    V16 bqm;
#pragma unroll
    for (int j = 0; j < 8; ++j) bqm.u[j] = 0u;
    if (hh == 0) {
#pragma unroll
        for (int j = 0; j < 4; ++j)
            bqm.u[j] = qpk[(size_t)(b * 4 + j) * NTOK + n];
    }

    const uint32_t* vrow32 =
        (const uint32_t*)(vh + (size_t)(b * CH + cbase + ln) * NTOK);
    const uint32_t* krow = kpk + (size_t)(b * 4) * NTOK;

    // permuted m offset for the K-tile A-matrix rows (loading lanes have hh==0)
    const int mrow = ln + (ln & 8);   // ln<8 -> ln ; ln>=8 -> ln+8

    float m_run = -3.0e38f;
    float l_run = 0.0f;
    v8f   accv  = {};
    const v8f cz = {};

    for (int m0 = 0; m0 < NTOK; m0 += 32) {
        // ---- K tiles as A-matrices (rows = permuted m, K = channel 0..7) ----
        V16 ak0, ak1;
#pragma unroll
        for (int j = 0; j < 8; ++j) { ak0.u[j] = 0u; ak1.u[j] = 0u; }
        if (hh == 0) {
#pragma unroll
            for (int j = 0; j < 4; ++j) {
                ak0.u[j] = krow[(size_t)j * NTOK + m0 + mrow];
                ak1.u[j] = krow[(size_t)j * NTOK + m0 + mrow + 8];
            }
        }

        // E half h, vgpr r:  e0[r] = P[m0+16h+r][n],  e1[r] = P[m0+16h+8+r][n]
        v8f e0 = __builtin_amdgcn_wmma_f32_16x16x32_f16(
            false, ak0.v, false, bqm.v, (short)0, cz, false, false);
        v8f e1 = __builtin_amdgcn_wmma_f32_16x16x32_f16(
            false, ak1.v, false, bqm.v, (short)0, cz, false, false);

        // ---- online softmax stats (row n = lane&15; combine halves in VALU)
        float mx = m_run;
#pragma unroll
        for (int r = 0; r < 8; ++r) mx = fmaxf(mx, fmaxf(e0[r], e1[r]));
        mx = fmaxf(mx, swap16f(mx));

        float p0[8], p1[8];
        float s = 0.0f;
#pragma unroll
        for (int r = 0; r < 8; ++r) {
            p0[r] = __expf(e0[r] - mx);
            p1[r] = __expf(e1[r] - mx);
            s += p0[r] + p1[r];
        }
        s += swap16f(s);

        const float scale = __expf(m_run - mx);
        l_run = l_run * scale + s;
        m_run = mx;
#pragma unroll
        for (int r = 0; r < 8; ++r) accv[r] *= scale;

        // ---- Pt as B-matrix (K = m-m0 in 0..31): all packs are lane-local ----
        V16 bp;
#pragma unroll
        for (int j = 0; j < 4; ++j) {
            bp.u[j]     = pkh(p0[2 * j], p0[2 * j + 1]);   // K = 16h + 2j,2j+1
            bp.u[4 + j] = pkh(p1[2 * j], p1[2 * j + 1]);   // K = 16h + 8+2j,..
        }

        // ---- V tile as A-matrix (rows = channel, K = m-m0, fully valid) ----
        V16 av;
        const uint32_t* vr = vrow32 + (m0 >> 1) + (hh << 3);
#pragma unroll
        for (int j = 0; j < 4; ++j) {
            av.u[j]     = vr[j];
            av.u[4 + j] = vr[4 + j];
        }

        accv = __builtin_amdgcn_wmma_f32_16x16x32_f16(
            false, av.v, false, bp.v, (short)0, accv, false, false);
    }

    // ---- epilogue: gamma * (acc / l) + x ----
    const float rl = 1.0f / l_run;
#pragma unroll
    for (int r = 0; r < 8; ++r) {
        const int c = cbase + r + (hh << 3);
        const size_t o = (size_t)(b * CH + c) * NTOK + n;
        out[o] = gamma * accv[r] * rl + x[o];
    }
}

// ---------------------------------------------------------------------------
extern "C" void kernel_launch(void* const* d_in, const int* in_sizes, int n_in,
                              void* d_out, int out_size, void* d_ws, size_t ws_size,
                              hipStream_t stream) {
    (void)in_sizes; (void)n_in; (void)out_size; (void)ws_size;
    const float* x     = (const float*)d_in[0];
    const float* ctx   = (const float*)d_in[1];
    const float* Wq    = (const float*)d_in[2];
    const float* bq    = (const float*)d_in[3];
    const float* Wk    = (const float*)d_in[4];
    const float* bk    = (const float*)d_in[5];
    const float* Wv    = (const float*)d_in[6];
    const float* bv    = (const float*)d_in[7];
    const float* gamma = (const float*)d_in[8];
    float* out = (float*)d_out;

    // workspace layout
    uint32_t* qpk = (uint32_t*)d_ws;                       // B*4*N u32
    uint32_t* kpk = qpk + (size_t)BATCH * 4 * NTOK;        // B*4*N u32
    uint16_t* vh  = (uint16_t*)(kpk + (size_t)BATCH * 4 * NTOK); // B*64*N f16

    ca_proj_kernel<<<(BATCH * NTOK) / 256, 256, 0, stream>>>(
        x, ctx, Wq, bq, Wk, bk, Wv, bv, qpk, kpk, vh);

    ca_attn_kernel<<<BATCH * (NTOK / 16), 128, 0, stream>>>(
        qpk, kpk, vh, x, gamma, out);
}
